// RNNModel_49357764165997
// MI455X (gfx1250) — matrix operand order repaired and checked by reading, not compiled, not run on hardware
//
#include <hip/hip_runtime.h>
#include <hip/hip_bf16.h>

#define B_   128
#define S_   512
#define V_   2000
#define EMB_ 96
#define D_   128
#define H_   512
#define ENC_ 256

// d_out offsets (float elements), outputs concatenated in reference return order
#define O_CATP   ((size_t)0)          // [B,S,1]
#define O_NONCAT ((size_t)65536)      // [B,S,32]
#define O_LOGITS ((size_t)2162688)    // [B,S,2000]
#define O_H1     ((size_t)133234688)  // [2,B,512]
#define O_C1     ((size_t)133365760)  // [2,B,512]
#define O_H2     ((size_t)133496832)  // [2,B,128]
#define O_C2     ((size_t)133529600)  // [2,B,128]

typedef __attribute__((ext_vector_type(8)))  __bf16 bf16x8;
typedef __attribute__((ext_vector_type(16))) __bf16 bf16x16;
typedef __attribute__((ext_vector_type(8)))  float  v8f;

union Frag { v8f v; float f[8]; };

__device__ __forceinline__ unsigned short f2bf(float x) {
    unsigned int u = __float_as_uint(x);
    u += 0x7fffu + ((u >> 16) & 1u);           // round-to-nearest-even
    return (unsigned short)(u >> 16);
}
__device__ __forceinline__ float bf2f(unsigned short h) {
    return __uint_as_float(((unsigned int)h) << 16);
}
__device__ __forceinline__ float sigf(float x) {
    return __builtin_amdgcn_rcpf(1.0f + __expf(-x));
}
__device__ __forceinline__ float tanhf_fast(float x) {
    x = fminf(fmaxf(x, -15.0f), 15.0f);
    float e = __expf(2.0f * x);
    return (e - 1.0f) * __builtin_amdgcn_rcpf(e + 1.0f);
}

// A fragment: 16x32 bf16, row-major source (row stride ld elems).
// Lane l holds row m=l&15; K blocks per ISA table: lanes<16 -> K {0..7,16..23}, lanes>=16 -> {8..15,24..31}.
__device__ __forceinline__ bf16x16 load_frag_a(const unsigned short* base, int ld, int kc, int lane) {
    const int m  = lane & 15;
    const int k0 = kc + ((lane >> 4) << 3);
    bf16x8 a0 = *reinterpret_cast<const bf16x8*>(base + (size_t)m * ld + k0);
    bf16x8 a1 = *reinterpret_cast<const bf16x8*>(base + (size_t)m * ld + k0 + 16);
    return __builtin_shufflevector(a0, a1, 0,1,2,3,4,5,6,7,8,9,10,11,12,13,14,15);
}
// B fragment: 32x16 bf16 = W^T tile; W stored [N,K] row-major (ld=K).
// Lane l holds column n=ncol+(l&15); K blocks per ISA table: lanes<16 -> K {0..7,16..23}, lanes>=16 -> {8..15,24..31}.
__device__ __forceinline__ bf16x16 load_frag_b(const unsigned short* W, int ld, int ncol, int kc, int lane) {
    const int n  = ncol + (lane & 15);
    const int k0 = kc + ((lane >> 4) << 3);
    bf16x8 b0 = *reinterpret_cast<const bf16x8*>(W + (size_t)n * ld + k0);
    bf16x8 b1 = *reinterpret_cast<const bf16x8*>(W + (size_t)n * ld + k0 + 16);
    return __builtin_shufflevector(b0, b1, 0,1,2,3,4,5,6,7,8,9,10,11,12,13,14,15);
}
__device__ __forceinline__ v8f wmma_bf16(bf16x16 a, bf16x16 b, v8f c) {
    return __builtin_amdgcn_wmma_f32_16x16x32_bf16(false, a, false, b, (short)0, c, false, false);
}

// ---------------------------------------------------------------------------
// One LSTM layer step for a 16-row batch block.
// NT = column-pairs per wave (2 for Hout=512 with 16 waves, 1 for Hout=128 with 8 waves).
// Wave w owns gate columns {g*Hout + w*16 + p*(Hout/NT)} for g=i,f,g,o -> pointwise LSTM
// math stays entirely inside the WMMA accumulator fragments; c never leaves registers.
// ---------------------------------------------------------------------------
template<int NT>
__device__ __forceinline__ void lstm_layer_step(
    int w, int lane,
    const unsigned short* __restrict__ xbase, int xld, int KX,
    const unsigned short* __restrict__ hprev, int hld, int KH,
    const unsigned short* __restrict__ Wih,
    const unsigned short* __restrict__ Whh,
    const float* __restrict__ bias, int Hout,
    unsigned short* hout, int hold,
    Frag* cfr)
{
    const int  pstep  = Hout / NT;
    const bool active = w < (Hout / (16 * NT));
    Frag acc[4 * NT];
    if (active) {
#pragma unroll
        for (int g = 0; g < 4; ++g)
#pragma unroll
            for (int p = 0; p < NT; ++p) {
                float bv = bias[g * Hout + w * 16 + p * pstep + (lane & 15)];
#pragma unroll
                for (int r = 0; r < 8; ++r) acc[g * NT + p].f[r] = bv;
            }
        for (int kc = 0; kc < KX; kc += 32) {
            bf16x16 a = load_frag_a(xbase, xld, kc, lane);
#pragma unroll
            for (int g = 0; g < 4; ++g)
#pragma unroll
                for (int p = 0; p < NT; ++p) {
                    bf16x16 b = load_frag_b(Wih, KX, g * Hout + w * 16 + p * pstep, kc, lane);
                    acc[g * NT + p].v = wmma_bf16(a, b, acc[g * NT + p].v);
                }
        }
        for (int kc = 0; kc < KH; kc += 32) {
            bf16x16 a = load_frag_a(hprev, hld, kc, lane);
#pragma unroll
            for (int g = 0; g < 4; ++g)
#pragma unroll
                for (int p = 0; p < NT; ++p) {
                    bf16x16 b = load_frag_b(Whh, KH, g * Hout + w * 16 + p * pstep, kc, lane);
                    acc[g * NT + p].v = wmma_bf16(a, b, acc[g * NT + p].v);
                }
        }
    }
    __syncthreads();                     // all waves done reading h(t-1)
    if (active) {
#pragma unroll
        for (int p = 0; p < NT; ++p)
#pragma unroll
            for (int r = 0; r < 8; ++r) {
                float iv = sigf(acc[0 * NT + p].f[r]);
                float fv = sigf(acc[1 * NT + p].f[r]);
                float gv = tanhf_fast(acc[2 * NT + p].f[r]);
                float ov = sigf(acc[3 * NT + p].f[r]);
                float cn = fv * cfr[p].f[r] + iv * gv;
                cfr[p].f[r] = cn;
                float hv = ov * tanhf_fast(cn);
                int m = r + ((lane >> 4) << 3);
                hout[(size_t)m * hold + w * 16 + p * pstep + (lane & 15)] = f2bf(hv);
            }
    }
    __syncthreads();                     // h(t) visible
}

// out[16,Hout] = relu(x[16,KX] @ W^T + b), wave w computes tile w (w < Hout/16)
__device__ __forceinline__ void fc_relu(
    int w, int lane,
    const unsigned short* __restrict__ x, int xld, int KX,
    const unsigned short* __restrict__ W, const float* __restrict__ b,
    int Hout, unsigned short* out, int old_)
{
    const bool active = w < (Hout / 16);
    Frag acc;
    if (active) {
        int ncol = w * 16;
        float bv = b[ncol + (lane & 15)];
#pragma unroll
        for (int r = 0; r < 8; ++r) acc.f[r] = bv;
        for (int kc = 0; kc < KX; kc += 32) {
            bf16x16 a = load_frag_a(x, xld, kc, lane);
            bf16x16 bb = load_frag_b(W, KX, ncol, kc, lane);
            acc.v = wmma_bf16(a, bb, acc.v);
        }
    }
    __syncthreads();
    if (active) {
#pragma unroll
        for (int r = 0; r < 8; ++r) {
            int m = r + ((lane >> 4) << 3);
            out[(size_t)m * old_ + w * 16 + (lane & 15)] = f2bf(fmaxf(acc.f[r], 0.0f));
        }
    }
    __syncthreads();
}

// ---------------------------------------------------------------------------
// Kernels
// ---------------------------------------------------------------------------
__global__ void cvt_kernel(const float* __restrict__ src, unsigned short* __restrict__ dst, int n) {
    int i = blockIdx.x * blockDim.x + threadIdx.x;
    if (i < n) dst[i] = f2bf(src[i]);
}

__global__ void embed_kernel(const int* __restrict__ xcat, const float* __restrict__ xrest,
                             const float* __restrict__ emb, unsigned short* __restrict__ xin) {
    int bs = blockIdx.x;                 // b*S+s
    int d  = threadIdx.x;                // 0..127
    float v;
    if (d < EMB_) {
        float s = 0.0f;
#pragma unroll
        for (int k = 0; k < 4; ++k)
            s += emb[(size_t)xcat[(size_t)bs * 4 + k] * EMB_ + d];
        v = s * 0.25f;
    } else {
        v = xrest[(size_t)bs * 32 + (d - EMB_)];
    }
    xin[(size_t)bs * D_ + d] = f2bf(v);
}

// Persistent encoder->FC->decoder kernel: 8 WGs x 512 threads (16 waves).
// WG owns batch rows [blockIdx.x*16, +16); all layer/time dependencies are WG-local.
__global__ __launch_bounds__(512) void seq2seq_kernel(
    const unsigned short* __restrict__ x_in,
    const unsigned short* __restrict__ W10i, const unsigned short* __restrict__ W10h, const float* __restrict__ b10,
    const unsigned short* __restrict__ W11i, const unsigned short* __restrict__ W11h, const float* __restrict__ b11,
    const unsigned short* __restrict__ Wf1,  const float* __restrict__ bf1,
    const unsigned short* __restrict__ Wf2,  const float* __restrict__ bf2,
    const unsigned short* __restrict__ W20i, const unsigned short* __restrict__ W20h, const float* __restrict__ b20,
    const unsigned short* __restrict__ W21i, const unsigned short* __restrict__ W21h, const float* __restrict__ b21,
    const float* __restrict__ h1_in, const float* __restrict__ c1_in,
    const float* __restrict__ h2_in, const float* __restrict__ c2_in,
    unsigned short* __restrict__ dec_out, float* __restrict__ out)
{
    __shared__ __align__(16) unsigned short sh0[16][512];   // h of encoder layer0 / enc / etc.
    __shared__ __align__(16) unsigned short sh1[16][512];   // h of encoder layer1 / decoder buffers

    const int tid  = threadIdx.x;
    const int w    = tid >> 5;
    const int lane = tid & 31;
    const int row0 = blockIdx.x * 16;

    // ---- encoder initial state ----
    for (int idx = tid; idx < 16 * 512; idx += 512) {
        int m = idx >> 9, col = idx & 511;
        sh0[m][col] = f2bf(h1_in[(size_t)(row0 + m) * H_ + col]);
        sh1[m][col] = f2bf(h1_in[(size_t)B_ * H_ + (size_t)(row0 + m) * H_ + col]);
    }
    Frag c10[2], c11[2];
#pragma unroll
    for (int p = 0; p < 2; ++p)
#pragma unroll
        for (int r = 0; r < 8; ++r) {
            int m = r + ((lane >> 4) << 3);
            int col = w * 16 + p * 256 + (lane & 15);
            c10[p].f[r] = c1_in[(size_t)(row0 + m) * H_ + col];
            c11[p].f[r] = c1_in[(size_t)B_ * H_ + (size_t)(row0 + m) * H_ + col];
        }
    __syncthreads();

    // ---- encoder: 512 steps, 2 stacked layers ----
    for (int t = 0; t < S_; ++t) {
        const unsigned short* xg = x_in + ((size_t)row0 * S_ + t) * D_;
        if (tid < 16 && (t + 1) < S_)
            __builtin_prefetch(x_in + ((size_t)(row0 + tid) * S_ + (t + 1)) * D_, 0, 3);
        lstm_layer_step<2>(w, lane, xg, S_ * D_, D_, &sh0[0][0], 512, H_,
                           W10i, W10h, b10, H_, &sh0[0][0], 512, c10);
        lstm_layer_step<2>(w, lane, &sh0[0][0], 512, H_, &sh1[0][0], 512, H_,
                           W11i, W11h, b11, H_, &sh1[0][0], 512, c11);
    }

    // ---- write encoder final states (h1a/h1b, c1a/c1b) ----
    for (int idx = tid; idx < 16 * 512; idx += 512) {
        int m = idx >> 9, col = idx & 511;
        out[O_H1 + (size_t)(row0 + m) * H_ + col] = bf2f(sh0[m][col]);
        out[O_H1 + (size_t)B_ * H_ + (size_t)(row0 + m) * H_ + col] = bf2f(sh1[m][col]);
    }
#pragma unroll
    for (int p = 0; p < 2; ++p)
#pragma unroll
        for (int r = 0; r < 8; ++r) {
            int m = r + ((lane >> 4) << 3);
            int col = w * 16 + p * 256 + (lane & 15);
            out[O_C1 + (size_t)(row0 + m) * H_ + col] = c10[p].f[r];
            out[O_C1 + (size_t)B_ * H_ + (size_t)(row0 + m) * H_ + col] = c11[p].f[r];
        }
    __syncthreads();

    // ---- fc1: enc[16,256] = relu(h1_last @ fc1_W^T + b), into sh0 cols 0..255 ----
    fc_relu(w, lane, &sh1[0][0], 512, H_, Wf1, bf1, ENC_, &sh0[0][0], 512);
    // ---- fc2: dec[16,128] = relu(enc @ fc2_W^T + b), into sh1 cols 0..127 (decoder input) ----
    fc_relu(w, lane, &sh0[0][0], 512, ENC_, Wf2, bf2, D_, &sh1[0][0], 512);

    // ---- decoder buffers inside sh1: inp[0..128) h20[128..256) h21[256..384) ----
    unsigned short* inp = &sh1[0][0];
    unsigned short* h20 = &sh1[0][128];
    unsigned short* h21 = &sh1[0][256];
    for (int idx = tid; idx < 16 * 128; idx += 512) {
        int m = idx >> 7, d = idx & 127;
        h20[(size_t)m * 512 + d] = f2bf(h2_in[(size_t)(row0 + m) * D_ + d]);
        h21[(size_t)m * 512 + d] = f2bf(h2_in[(size_t)B_ * D_ + (size_t)(row0 + m) * D_ + d]);
    }
    Frag c20, c21;
    if (w < 8) {
#pragma unroll
        for (int r = 0; r < 8; ++r) {
            int m = r + ((lane >> 4) << 3);
            int col = w * 16 + (lane & 15);
            c20.f[r] = c2_in[(size_t)(row0 + m) * D_ + col];
            c21.f[r] = c2_in[(size_t)B_ * D_ + (size_t)(row0 + m) * D_ + col];
        }
    }
    __syncthreads();

    // ---- decoder: 512 autoregressive steps, 2 stacked layers ----
    for (int t = 0; t < S_; ++t) {
        lstm_layer_step<1>(w, lane, inp, 512, D_, h20, 512, D_,
                           W20i, W20h, b20, D_, h20, 512, &c20);
        lstm_layer_step<1>(w, lane, h20, 512, D_, h21, 512, D_,
                           W21i, W21h, b21, D_, h21, 512, &c21);
        // out_t = h21 -> store + becomes next input
        for (int idx = tid; idx < 16 * 128; idx += 512) {
            int m = idx >> 7, d = idx & 127;
            unsigned short hv = h21[(size_t)m * 512 + d];
            inp[(size_t)m * 512 + d] = hv;
            dec_out[((size_t)(row0 + m) * S_ + t) * D_ + d] = hv;
        }
        __syncthreads();
    }

    // ---- write decoder final states ----
    for (int idx = tid; idx < 16 * 128; idx += 512) {
        int m = idx >> 7, d = idx & 127;
        out[O_H2 + (size_t)(row0 + m) * D_ + d] = bf2f(h20[(size_t)m * 512 + d]);
        out[O_H2 + (size_t)B_ * D_ + (size_t)(row0 + m) * D_ + d] = bf2f(h21[(size_t)m * 512 + d]);
    }
    if (w < 8) {
#pragma unroll
        for (int r = 0; r < 8; ++r) {
            int m = r + ((lane >> 4) << 3);
            int col = w * 16 + (lane & 15);
            out[O_C2 + (size_t)(row0 + m) * D_ + col] = c20.f[r];
            out[O_C2 + (size_t)B_ * D_ + (size_t)(row0 + m) * D_ + col] = c21.f[r];
        }
    }
}

// cat_logits = dec_out[:, :, :96] @ cat_W^T + cat_b.  M=65536, N=2000 (125 tiles), K=96 (3 chunks).
__global__ __launch_bounds__(256) void logits_kernel(
    const unsigned short* __restrict__ X, const unsigned short* __restrict__ Wc,
    const float* __restrict__ bc, float* __restrict__ out)
{
    const int w = threadIdx.x >> 5, lane = threadIdx.x & 31;
    const int mtile = blockIdx.x;
    const int ntile = blockIdx.y * 8 + w;
    if (ntile >= V_ / 16) return;
    const unsigned short* a_base = X + (size_t)mtile * 16 * D_;
    const int ncol = ntile * 16;
    Frag acc;
    float bv = bc[ncol + (lane & 15)];
#pragma unroll
    for (int r = 0; r < 8; ++r) acc.f[r] = bv;
#pragma unroll
    for (int kc = 0; kc < EMB_; kc += 32) {
        bf16x16 a = load_frag_a(a_base, D_, kc, lane);
        bf16x16 b = load_frag_b(Wc, EMB_, ncol, kc, lane);
        acc.v = wmma_bf16(a, b, acc.v);
    }
#pragma unroll
    for (int r = 0; r < 8; ++r) {
        int m = r + ((lane >> 4) << 3);
        out[(size_t)(mtile * 16 + m) * V_ + ncol + (lane & 15)] = acc.f[r];
    }
}

__global__ __launch_bounds__(256) void argmax_kernel(const float* __restrict__ logits,
                                                     float* __restrict__ pred) {
    int row  = blockIdx.x * 8 + (threadIdx.x >> 5);
    int lane = threadIdx.x & 31;
    const float* p = logits + (size_t)row * V_;
    float best = -3.4e38f; int bi = 0;
    for (int i = lane; i < V_; i += 32) {
        float v = p[i];
        if (v > best) { best = v; bi = i; }
    }
#pragma unroll
    for (int off = 16; off; off >>= 1) {
        float ob = __shfl_xor(best, off, 32);
        int   oi = __shfl_xor(bi, off, 32);
        if (ob > best || (ob == best && oi < bi)) { best = ob; bi = oi; }
    }
    if (lane == 0) pred[row] = (float)bi;
}

__global__ void noncat_kernel(const unsigned short* __restrict__ dec_out, float* __restrict__ out) {
    for (size_t idx = (size_t)blockIdx.x * blockDim.x + threadIdx.x;
         idx < (size_t)B_ * S_ * 32; idx += (size_t)gridDim.x * blockDim.x) {
        size_t bs = idx >> 5;
        int d = (int)(idx & 31);
        out[idx] = bf2f(dec_out[bs * D_ + EMB_ + d]);
    }
}

// ---------------------------------------------------------------------------
extern "C" void kernel_launch(void* const* d_in, const int* in_sizes, int n_in,
                              void* d_out, int out_size, void* d_ws, size_t ws_size,
                              hipStream_t stream)
{
    const int*   x_cat   = (const int*)  d_in[0];
    const float* x_rest  = (const float*)d_in[1];
    const float* h1      = (const float*)d_in[2];
    const float* c1      = (const float*)d_in[3];
    const float* h2      = (const float*)d_in[4];
    const float* c2      = (const float*)d_in[5];
    const float* emb     = (const float*)d_in[6];
    const float* l1_Wih0 = (const float*)d_in[7];
    const float* l1_Whh0 = (const float*)d_in[8];
    const float* l1_b0   = (const float*)d_in[9];
    const float* l1_Wih1 = (const float*)d_in[10];
    const float* l1_Whh1 = (const float*)d_in[11];
    const float* l1_b1   = (const float*)d_in[12];
    const float* fc1_W   = (const float*)d_in[13];
    const float* fc1_b   = (const float*)d_in[14];
    const float* fc2_W   = (const float*)d_in[15];
    const float* fc2_b   = (const float*)d_in[16];
    const float* l2_Wih0 = (const float*)d_in[17];
    const float* l2_Whh0 = (const float*)d_in[18];
    const float* l2_b0   = (const float*)d_in[19];
    const float* l2_Wih1 = (const float*)d_in[20];
    const float* l2_Whh1 = (const float*)d_in[21];
    const float* l2_b1   = (const float*)d_in[22];
    const float* cat_W   = (const float*)d_in[23];
    const float* cat_b   = (const float*)d_in[24];
    float* out = (float*)d_out;

    unsigned char* ws = (unsigned char*)d_ws;
    size_t off = 0;
    auto alloc = [&](size_t bytes) -> unsigned short* {
        unsigned short* p = (unsigned short*)(ws + off);
        off += (bytes + 255) & ~(size_t)255;
        return p;
    };
    unsigned short* x_in  = alloc((size_t)B_ * S_ * D_ * 2);
    unsigned short* dec   = alloc((size_t)B_ * S_ * D_ * 2);
    unsigned short* w10i  = alloc((size_t)4 * H_ * D_   * 2);
    unsigned short* w10h  = alloc((size_t)4 * H_ * H_   * 2);
    unsigned short* w11i  = alloc((size_t)4 * H_ * H_   * 2);
    unsigned short* w11h  = alloc((size_t)4 * H_ * H_   * 2);
    unsigned short* wf1   = alloc((size_t)ENC_ * H_     * 2);
    unsigned short* wf2   = alloc((size_t)D_ * ENC_     * 2);
    unsigned short* w20i  = alloc((size_t)4 * D_ * D_   * 2);
    unsigned short* w20h  = alloc((size_t)4 * D_ * D_   * 2);
    unsigned short* w21i  = alloc((size_t)4 * D_ * D_   * 2);
    unsigned short* w21h  = alloc((size_t)4 * D_ * D_   * 2);
    unsigned short* wcat  = alloc((size_t)V_ * EMB_     * 2);

    auto cvt = [&](const float* s, unsigned short* d, int n) {
        cvt_kernel<<<(n + 255) / 256, 256, 0, stream>>>(s, d, n);
    };
    cvt(l1_Wih0, w10i, 4 * H_ * D_);
    cvt(l1_Whh0, w10h, 4 * H_ * H_);
    cvt(l1_Wih1, w11i, 4 * H_ * H_);
    cvt(l1_Whh1, w11h, 4 * H_ * H_);
    cvt(fc1_W,   wf1,  ENC_ * H_);
    cvt(fc2_W,   wf2,  D_ * ENC_);
    cvt(l2_Wih0, w20i, 4 * D_ * D_);
    cvt(l2_Whh0, w20h, 4 * D_ * D_);
    cvt(l2_Wih1, w21i, 4 * D_ * D_);
    cvt(l2_Whh1, w21h, 4 * D_ * D_);
    cvt(cat_W,   wcat, V_ * EMB_);

    embed_kernel<<<B_ * S_, 128, 0, stream>>>(x_cat, x_rest, emb, x_in);

    seq2seq_kernel<<<B_ / 16, 512, 0, stream>>>(
        x_in,
        w10i, w10h, l1_b0, w11i, w11h, l1_b1,
        wf1, fc1_b, wf2, fc2_b,
        w20i, w20h, l2_b0, w21i, w21h, l2_b1,
        h1, c1, h2, c2,
        dec, out);

    logits_kernel<<<dim3((B_ * S_) / 16, 16), 256, 0, stream>>>(dec, wcat, cat_b, out + O_LOGITS);
    argmax_kernel<<<(B_ * S_) / 8, 256, 0, stream>>>(out + O_LOGITS, out + O_CATP);
    noncat_kernel<<<8192, 256, 0, stream>>>(dec, out + O_NONCAT);
}
